// Net_18064632447621
// MI455X (gfx1250) — matrix operand order repaired and checked by reading, not compiled, not run on hardware
//
#include <hip/hip_runtime.h>
#include <hip/hip_bf16.h>
#include <math.h>

typedef __attribute__((ext_vector_type(16))) _Float16 v16h;
typedef __attribute__((ext_vector_type(8)))  _Float16 v8h;
typedef __attribute__((ext_vector_type(8)))  float    v8f;

#define NB    2
#define CENC  256
#define HENC  32
#define IMG   128
#define DD    4096
#define HEADS 8
#define FEAT  512
#define LL    64
#define TT    128
#define KNN   6
#define SLOPE 0.2f

// ---------------- encoder conv: 3->256, k=4, stride=4, VALID, relu ----------------
__global__ void k_enc_conv(const float* __restrict__ in, const float* __restrict__ w,
                           const float* __restrict__ b, float* __restrict__ out) {
  int idx = blockIdx.x * blockDim.x + threadIdx.x;              // NB*256*32*32 = 524288
  if (idx >= NB * CENC * HENC * HENC) return;
  int x = idx & 31, y = (idx >> 5) & 31, co = (idx >> 10) & 255, n = idx >> 18;
  float acc = b[co];
  for (int ci = 0; ci < 3; ++ci)
    for (int ky = 0; ky < 4; ++ky)
      for (int kx = 0; kx < 4; ++kx)
        acc += in[((n * 3 + ci) * IMG + y * 4 + ky) * IMG + x * 4 + kx] *
               w[((co * 3 + ci) * 4 + ky) * 4 + kx];
  out[idx] = fmaxf(acc, 0.f);
}

// ---------------- patch2feat: (n,256,32,32) -> (n,64,4096) ----------------
__global__ void k_patch2feat(const float* __restrict__ enc, float* __restrict__ f) {
  int idx = blockIdx.x * blockDim.x + threadIdx.x;              // NB*64*4096 = 524288
  if (idx >= NB * LL * DD) return;
  int d = idx & 4095, l = (idx >> 12) & 63, n = idx >> 18;
  int c = d >> 4, py = (d >> 2) & 3, px = d & 3;
  int ph = l >> 3, pw = l & 7;
  f[idx] = enc[((n * CENC + c) * HENC + ph * 4 + py) * HENC + pw * 4 + px];
}

// ---------------- normalized concat [cp;sp] -> allv (NB*128 rows) ----------------
__global__ void k_norm_allv(const float* __restrict__ cp, const float* __restrict__ sp,
                            float* __restrict__ allv) {
  int row = blockIdx.x;                                         // 0..NB*TT-1
  int n = row >> 7, t = row & 127;
  const float* src = (t < LL) ? (cp + (size_t)(n * LL + t) * DD)
                              : (sp + (size_t)(n * LL + t - LL) * DD);
  float* dst = allv + (size_t)row * DD;
  __shared__ float red[256];
  float ss = 0.f;
  for (int d = threadIdx.x; d < DD; d += 256) { float v = src[d]; ss += v * v; }
  red[threadIdx.x] = ss; __syncthreads();
  for (int s = 128; s > 0; s >>= 1) {
    if ((int)threadIdx.x < s) red[threadIdx.x] += red[threadIdx.x + s];
    __syncthreads();
  }
  float inv = 1.f / (sqrtf(red[0]) + 1e-12f);
  for (int d = threadIdx.x; d < DD; d += 256) dst[d] = src[d] * inv;
}

// ---------------- normalized content rows from upd -> un (NB*64 rows) ----------------
__global__ void k_norm_ucp(const float* __restrict__ upd, float* __restrict__ un) {
  int row = blockIdx.x;                                         // 0..NB*LL-1
  int n = row >> 6, t = row & 63;
  const float* src = upd + (size_t)(n * TT + t) * DD;
  float* dst = un + (size_t)row * DD;
  __shared__ float red[256];
  float ss = 0.f;
  for (int d = threadIdx.x; d < DD; d += 256) { float v = src[d]; ss += v * v; }
  red[threadIdx.x] = ss; __syncthreads();
  for (int s = 128; s > 0; s >>= 1) {
    if ((int)threadIdx.x < s) red[threadIdx.x] += red[threadIdx.x + s];
    __syncthreads();
  }
  float inv = 1.f / (sqrtf(red[0]) + 1e-12f);
  for (int d = threadIdx.x; d < DD; d += 256) dst[d] = src[d] * inv;
}

// ---------------- pairwise similarity: one wave per (n,i,j) ----------------
__global__ void k_sim(const float* __restrict__ v, float* __restrict__ sim, int Tt) {
  int gid = blockIdx.x * blockDim.x + threadIdx.x;
  int wid = gid >> 5, lane = threadIdx.x & 31;
  if (wid >= NB * Tt * Tt) return;
  int j = wid % Tt, i = (wid / Tt) % Tt, n = wid / (Tt * Tt);
  const float* a = v + (size_t)(n * Tt + i) * DD;
  const float* b = v + (size_t)(n * Tt + j) * DD;
  float acc = 0.f;
  for (int d = lane; d < DD; d += 32) acc += a[d] * b[d];
  for (int off = 16; off; off >>= 1) acc += __shfl_xor(acc, off, 32);
  if (lane == 0) sim[wid] = acc;
}

// ---------------- top-6 neighbor selection (stable, lower index wins ties) ----------------
__global__ void k_knn(const float* __restrict__ sim, int* __restrict__ idx,
                      int Tt, int maskCross) {
  int r = blockIdx.x * blockDim.x + threadIdx.x;
  if (r >= NB * Tt) return;
  int i = r % Tt;
  const float* row = sim + (size_t)r * Tt;
  int half = Tt >> 1;
  bool iblk = i < half;
  unsigned long long lo = 0ull, hi = 0ull;
  for (int s = 0; s < KNN; ++s) {
    float best = -3.0e38f; int bj = 0;
    for (int j = 0; j < Tt; ++j) {
      if (maskCross) { bool jblk = j < half; if (jblk == iblk) continue; }
      bool used = (j < 64) ? ((lo >> j) & 1ull) : ((hi >> (j - 64)) & 1ull);
      if (used) continue;
      float v = row[j];
      if (v > best) { best = v; bj = j; }
    }
    idx[r * KNN + s] = bj;
    if (bj < 64) lo |= 1ull << bj; else hi |= 1ull << (bj - 64);
  }
}

// ---------------- f32 -> f16 packers ----------------
__global__ void k_pack_nodes16(const float* __restrict__ cp, const float* __restrict__ sp,
                               _Float16* __restrict__ out) {
  int idx = blockIdx.x * blockDim.x + threadIdx.x;              // NB*128*4096
  if (idx >= NB * TT * DD) return;
  int d = idx & 4095, t = (idx >> 12) & 127, n = idx >> 19;
  float v = (t < LL) ? cp[(size_t)(n * LL + t) * DD + d]
                     : sp[(size_t)(n * LL + t - LL) * DD + d];
  out[idx] = (_Float16)v;
}

__global__ void k_pack_ucp16(const float* __restrict__ upd, _Float16* __restrict__ out) {
  int idx = blockIdx.x * blockDim.x + threadIdx.x;              // NB*64*4096
  if (idx >= NB * LL * DD) return;
  int d = idx & 4095, t = (idx >> 12) & 63, n = idx >> 18;
  out[idx] = (_Float16)upd[(size_t)(n * TT + t) * DD + d];
}

__global__ void k_f32_to_f16(const float* __restrict__ src, _Float16* __restrict__ dst, int count) {
  int idx = blockIdx.x * blockDim.x + threadIdx.x;
  if (idx < count) dst[idx] = (_Float16)src[idx];
}

// ---------------- weight convert + transpose: (K x N) f32 -> (N x K) f16 ----------------
__global__ void k_transpose_f16(const float* __restrict__ src, _Float16* __restrict__ dst) {
  __shared__ _Float16 tile[32][33];
  int bx = blockIdx.x, by = blockIdx.y;                         // bx over N/32, by over K/32
  int tx = threadIdx.x, ty = threadIdx.y;                       // 32 x 8
  for (int r = 0; r < 4; ++r) {
    int k = by * 32 + ty + r * 8;
    tile[ty + r * 8][tx] = (_Float16)src[(size_t)k * DD + bx * 32 + tx];
  }
  __syncthreads();
  for (int r = 0; r < 4; ++r) {
    int nrow = bx * 32 + ty + r * 8;
    dst[(size_t)nrow * DD + by * 32 + tx] = tile[tx][ty + r * 8];
  }
}

// ---------------- WMMA GEMM: C(MxN) = A(MxK,f16) * Bt(NxK,f16)^T + bias[N] ----------------
// one wave -> one 16x16 C tile; K-step 32 via v_wmma_f32_16x16x32_f16
__global__ void __launch_bounds__(128)
k_gemm_wmma(const _Float16* __restrict__ A, const _Float16* __restrict__ Bt,
            const float* __restrict__ bias, float* __restrict__ C,
            int M, int N, int K) {
  int lane  = threadIdx.x;                 // 0..31
  int ntile = blockIdx.x * 4 + threadIdx.y;
  int mtile = blockIdx.y;
  int mloc  = lane & 15;
  int hi    = lane >> 4;
  const _Float16* arow = A  + (size_t)(mtile * 16 + mloc) * K;  // row m of A
  const _Float16* brow = Bt + (size_t)(ntile * 16 + mloc) * K;  // col n of B
  v8f acc = {};
  for (int kb = 0; kb < K; kb += 32) {
    // A 16-bit layout: lane(hi=0): v0..3 = K 0..7, v4..7 = K 16..23 ; hi=1: +8
    v8h a0 = *(const v8h*)(arow + kb + hi * 8);
    v8h a1 = *(const v8h*)(arow + kb + 16 + hi * 8);
    // B 16-bit layout: same as A: hi=0: v0..3 = K 0..7, v4..7 = K 16..23 ; hi=1: +8
    v8h b0 = *(const v8h*)(brow + kb + hi * 8);
    v8h b1 = *(const v8h*)(brow + kb + 16 + hi * 8);
    v16h af, bf;
#pragma unroll
    for (int p = 0; p < 8; ++p) { af[p] = a0[p]; af[p + 8] = a1[p]; bf[p] = b0[p]; bf[p + 8] = b1[p]; }
    acc = __builtin_amdgcn_wmma_f32_16x16x32_f16(false, af, false, bf, (short)0, acc, false, false);
  }
  float bv = bias ? bias[ntile * 16 + mloc] : 0.f;
#pragma unroll
  for (int r = 0; r < 8; ++r) {
    int row = mtile * 16 + hi * 8 + r;     // C/D layout: vgpr r, lanes 16-31 -> M+8
    C[(size_t)row * N + ntile * 16 + mloc] = acc[r] + bv;
  }
}

// ---------------- GATv2 attention over the 6 KNN neighbors ----------------
__global__ void k_gat(const float* __restrict__ GL, const float* __restrict__ GR,
                      const float* __restrict__ aw, const int* __restrict__ idx,
                      float* __restrict__ out, int Tt) {
  int i = blockIdx.x, n = blockIdx.y;
  int h = threadIdx.x >> 5, lane = threadIdx.x & 31;
  __shared__ float esh[HEADS][KNN];
  int nb[KNN];
#pragma unroll
  for (int k = 0; k < KNN; ++k) nb[k] = idx[(n * Tt + i) * KNN + k];
  const float* gli = GL + (size_t)(n * Tt + i) * DD + h * FEAT;
  const float* ah  = aw + h * FEAT;
  for (int k = 0; k < KNN; ++k) {
    const float* grj = GR + (size_t)(n * Tt + nb[k]) * DD + h * FEAT;
    float p = 0.f;
    for (int f = lane; f < FEAT; f += 32) {
      float x = gli[f] + grj[f];
      x = x > 0.f ? x : SLOPE * x;
      p += ah[f] * x;
    }
    for (int off = 16; off; off >>= 1) p += __shfl_xor(p, off, 32);
    if (lane == 0) esh[h][k] = p;
  }
  __syncthreads();
  float e[KNN];
  float m = -3.0e38f;
#pragma unroll
  for (int k = 0; k < KNN; ++k) { e[k] = esh[h][k]; m = fmaxf(m, e[k]); }
  float s = 0.f;
#pragma unroll
  for (int k = 0; k < KNN; ++k) { e[k] = expf(e[k] - m); s += e[k]; }
  float inv = 1.f / s;
  float* oi = out + (size_t)(n * Tt + i) * DD + h * FEAT;
  for (int f = lane; f < FEAT; f += 32) {
    float acc = 0.f;
#pragma unroll
    for (int k = 0; k < KNN; ++k)
      acc += (e[k] * inv) * GR[(size_t)(n * Tt + nb[k]) * DD + h * FEAT + f];
    oi[f] = acc;
  }
}

// ---------------- feat2img + fold bias ----------------
__global__ void k_feat2img(const float* __restrict__ f, const float* __restrict__ fb,
                           float* __restrict__ img) {
  int idx = blockIdx.x * blockDim.x + threadIdx.x;              // NB*256*32*32
  if (idx >= NB * CENC * HENC * HENC) return;
  int x = idx & 31, y = (idx >> 5) & 31, c = (idx >> 10) & 255, n = idx >> 18;
  int l = (y >> 2) * 8 + (x >> 2);
  int d = c * 16 + (y & 3) * 4 + (x & 3);
  img[idx] = f[(size_t)(n * LL + l) * DD + d] + fb[(c * HENC + y) * HENC + x];
}

// ---------------- AdaIN: one block per (n,c) channel of 32x32 ----------------
__global__ void k_adain(const float* __restrict__ c, const float* __restrict__ s,
                        float* __restrict__ out) {
  int nc = blockIdx.x;                                          // NB*CENC
  const float* cp = c + (size_t)nc * 1024;
  const float* sp = s + (size_t)nc * 1024;
  __shared__ float r0[256], r1[256], r2[256], r3[256];
  float sc = 0.f, sc2 = 0.f, ss = 0.f, ss2 = 0.f;
  for (int p = threadIdx.x; p < 1024; p += 256) {
    float cv = cp[p], sv = sp[p];
    sc += cv; sc2 += cv * cv; ss += sv; ss2 += sv * sv;
  }
  r0[threadIdx.x] = sc; r1[threadIdx.x] = sc2; r2[threadIdx.x] = ss; r3[threadIdx.x] = ss2;
  __syncthreads();
  for (int st = 128; st > 0; st >>= 1) {
    if ((int)threadIdx.x < st) {
      r0[threadIdx.x] += r0[threadIdx.x + st];
      r1[threadIdx.x] += r1[threadIdx.x + st];
      r2[threadIdx.x] += r2[threadIdx.x + st];
      r3[threadIdx.x] += r3[threadIdx.x + st];
    }
    __syncthreads();
  }
  float cm = r0[0] * (1.f / 1024.f);
  float csd = sqrtf(fmaxf(r1[0] * (1.f / 1024.f) - cm * cm, 0.f));
  float sm = r2[0] * (1.f / 1024.f);
  float ssd = sqrtf(fmaxf(r3[0] * (1.f / 1024.f) - sm * sm, 0.f));
  float scale = ssd / (csd + 1e-5f);
  for (int p = threadIdx.x; p < 1024; p += 256)
    out[(size_t)nc * 1024 + p] = scale * (cp[p] - cm) + sm;
}

// ---------------- 3x3 conv, reflect pad 1, relu ----------------
__device__ __forceinline__ int refl(int i, int n) { return i < 0 ? -i : (i >= n ? 2 * n - 2 - i : i); }

__global__ void k_conv3(const float* __restrict__ in, const float* __restrict__ w,
                        const float* __restrict__ b, float* __restrict__ out,
                        int Cin, int Cout, int H, int W) {
  int idx = blockIdx.x * blockDim.x + threadIdx.x;
  int total = NB * Cout * H * W;
  if (idx >= total) return;
  int x = idx % W, y = (idx / W) % H, co = (idx / (W * H)) % Cout, n = idx / (W * H * Cout);
  float acc = b[co];
  const float* wc = w + (size_t)co * Cin * 9;
  for (int ci = 0; ci < Cin; ++ci) {
    const float* ip = in + (size_t)(n * Cin + ci) * H * W;
    const float* wp = wc + ci * 9;
    for (int dy = 0; dy < 3; ++dy) {
      int iy = refl(y + dy - 1, H);
      for (int dx = 0; dx < 3; ++dx) {
        int ix = refl(x + dx - 1, W);
        acc += ip[iy * W + ix] * wp[dy * 3 + dx];
      }
    }
  }
  out[idx] = fmaxf(acc, 0.f);
}

// ---------------- nearest 2x upsample ----------------
__global__ void k_up2(const float* __restrict__ in, float* __restrict__ out,
                      int C, int H, int W) {                    // H,W are INPUT dims
  int W2 = W * 2, H2 = H * 2;
  int idx = blockIdx.x * blockDim.x + threadIdx.x;
  int total = NB * C * H2 * W2;
  if (idx >= total) return;
  int x = idx % W2, y = (idx / W2) % H2, c = (idx / (W2 * H2)) % C, n = idx / (W2 * H2 * C);
  out[idx] = in[((size_t)(n * C + c) * H + (y >> 1)) * W + (x >> 1)];
}

// =====================================================================================
extern "C" void kernel_launch(void* const* d_in, const int* in_sizes, int n_in,
                              void* d_out, int out_size, void* d_ws, size_t ws_size,
                              hipStream_t stream) {
  (void)in_sizes; (void)n_in; (void)out_size; (void)ws_size;
  const float* style     = (const float*)d_in[0];
  const float* content   = (const float*)d_in[1];
  const float* enc_w     = (const float*)d_in[2];
  const float* enc_b     = (const float*)d_in[3];
  const float* wl1       = (const float*)d_in[4];
  const float* bl1       = (const float*)d_in[5];
  const float* wr1       = (const float*)d_in[6];
  const float* br1       = (const float*)d_in[7];
  const float* a1        = (const float*)d_in[8];
  const float* wl2       = (const float*)d_in[9];
  const float* bl2       = (const float*)d_in[10];
  const float* wr2       = (const float*)d_in[11];
  const float* br2       = (const float*)d_in[12];
  const float* a2        = (const float*)d_in[13];
  const float* wf        = (const float*)d_in[14];
  const float* fold_bias = (const float*)d_in[15];
  const float* d1w = (const float*)d_in[16]; const float* d1b = (const float*)d_in[17];
  const float* d2w = (const float*)d_in[18]; const float* d2b = (const float*)d_in[19];
  const float* d3w = (const float*)d_in[20]; const float* d3b = (const float*)d_in[21];
  const float* d4w = (const float*)d_in[22]; const float* d4b = (const float*)d_in[23];
  const float* d5w = (const float*)d_in[24]; const float* d5b = (const float*)d_in[25];

  char* ws = (char*)d_ws;
  size_t off = 0;
  auto alloc = [&](size_t bytes) -> char* {
    char* p = ws + off;
    off += (bytes + 255) & ~(size_t)255;
    return p;
  };
  _Float16* W16     = (_Float16*)alloc((size_t)DD * DD * 2);        // 33.5 MB, reused 5x
  float* ENC_S      = (float*)alloc((size_t)NB * CENC * 1024 * 4);
  float* ENC_C      = (float*)alloc((size_t)NB * CENC * 1024 * 4);
  float* CP         = (float*)alloc((size_t)NB * LL * DD * 4);
  float* SP         = (float*)alloc((size_t)NB * LL * DD * 4);
  float* ALLV       = (float*)alloc((size_t)NB * TT * DD * 4);      // reused as UN later
  float* SIM        = (float*)alloc((size_t)NB * TT * TT * 4);      // reused for sim2
  int*   IDX1       = (int*)alloc((size_t)NB * TT * KNN * 4);
  int*   IDX2       = (int*)alloc((size_t)NB * LL * KNN * 4);
  _Float16* NODES16 = (_Float16*)alloc((size_t)NB * TT * DD * 2);
  float* GL1        = (float*)alloc((size_t)NB * TT * DD * 4);
  float* GR1        = (float*)alloc((size_t)NB * TT * DD * 4);
  float* UPD        = (float*)alloc((size_t)NB * TT * DD * 4);
  _Float16* UCP16   = (_Float16*)alloc((size_t)NB * LL * DD * 2);
  float* GL2        = (float*)alloc((size_t)NB * LL * DD * 4);
  float* GR2        = (float*)alloc((size_t)NB * LL * DD * 4);
  float* FCP        = (float*)alloc((size_t)NB * LL * DD * 4);
  _Float16* FCP16   = (_Float16*)alloc((size_t)NB * LL * DD * 2);
  float* FCPW       = (float*)alloc((size_t)NB * LL * DD * 4);
  float* FOLD       = (float*)alloc((size_t)NB * CENC * 1024 * 4);
  float* XA         = (float*)alloc((size_t)NB * 64 * IMG * IMG * 4);
  float* XB         = (float*)alloc((size_t)NB * 64 * IMG * IMG * 4);

  const dim3 tb(256);
  // 1) encoder
  k_enc_conv<<<2048, tb, 0, stream>>>(style, enc_w, enc_b, ENC_S);
  k_enc_conv<<<2048, tb, 0, stream>>>(content, enc_w, enc_b, ENC_C);
  // 2) tokens
  k_patch2feat<<<2048, tb, 0, stream>>>(ENC_S, SP);
  k_patch2feat<<<2048, tb, 0, stream>>>(ENC_C, CP);
  // 3) sim1 + knn1 (cross-block mask)
  k_norm_allv<<<NB * TT, tb, 0, stream>>>(CP, SP, ALLV);
  k_sim<<<(NB * TT * TT * 32) / 256, tb, 0, stream>>>(ALLV, SIM, TT);
  k_knn<<<1, tb, 0, stream>>>(SIM, IDX1, TT, 1);
  // 4) GAT layer 1 projections via WMMA
  k_pack_nodes16<<<4096, tb, 0, stream>>>(CP, SP, NODES16);
  k_transpose_f16<<<dim3(128, 128), dim3(32, 8), 0, stream>>>(wl1, W16);
  k_gemm_wmma<<<dim3(DD / 64, (NB * TT) / 16), dim3(32, 4), 0, stream>>>(NODES16, W16, bl1, GL1, NB * TT, DD, DD);
  k_transpose_f16<<<dim3(128, 128), dim3(32, 8), 0, stream>>>(wr1, W16);
  k_gemm_wmma<<<dim3(DD / 64, (NB * TT) / 16), dim3(32, 4), 0, stream>>>(NODES16, W16, br1, GR1, NB * TT, DD, DD);
  k_gat<<<dim3(TT, NB), tb, 0, stream>>>(GL1, GR1, a1, IDX1, UPD, TT);
  // 5) sim2 + knn2 (no mask, includes self)
  k_norm_ucp<<<NB * LL, tb, 0, stream>>>(UPD, ALLV);
  k_sim<<<(NB * LL * LL * 32) / 256, tb, 0, stream>>>(ALLV, SIM, LL);
  k_knn<<<1, tb, 0, stream>>>(SIM, IDX2, LL, 0);
  // 6) GAT layer 2
  k_pack_ucp16<<<2048, tb, 0, stream>>>(UPD, UCP16);
  k_transpose_f16<<<dim3(128, 128), dim3(32, 8), 0, stream>>>(wl2, W16);
  k_gemm_wmma<<<dim3(DD / 64, (NB * LL) / 16), dim3(32, 4), 0, stream>>>(UCP16, W16, bl2, GL2, NB * LL, DD, DD);
  k_transpose_f16<<<dim3(128, 128), dim3(32, 8), 0, stream>>>(wr2, W16);
  k_gemm_wmma<<<dim3(DD / 64, (NB * LL) / 16), dim3(32, 4), 0, stream>>>(UCP16, W16, br2, GR2, NB * LL, DD, DD);
  k_gat<<<dim3(LL, NB), tb, 0, stream>>>(GL2, GR2, a2, IDX2, FCP, LL);
  // 7) fold: fcp @ wf -> image + bias
  k_f32_to_f16<<<2048, tb, 0, stream>>>(FCP, FCP16, NB * LL * DD);
  k_transpose_f16<<<dim3(128, 128), dim3(32, 8), 0, stream>>>(wf, W16);
  k_gemm_wmma<<<dim3(DD / 64, (NB * LL) / 16), dim3(32, 4), 0, stream>>>(FCP16, W16, (const float*)nullptr, FCPW, NB * LL, DD, DD);
  k_feat2img<<<2048, tb, 0, stream>>>(FCPW, fold_bias, FOLD);
  // 8) adain
  k_adain<<<NB * CENC, tb, 0, stream>>>(FOLD, ENC_S, XA);
  // 9) decoder
  k_conv3<<<(NB * 128 * 32 * 32 + 255) / 256, tb, 0, stream>>>(XA, d1w, d1b, XB, 256, 128, 32, 32);
  k_conv3<<<(NB * 128 * 32 * 32 + 255) / 256, tb, 0, stream>>>(XB, d2w, d2b, XA, 128, 128, 32, 32);
  k_up2<<<(NB * 128 * 64 * 64 + 255) / 256, tb, 0, stream>>>(XA, XB, 128, 32, 32);
  k_conv3<<<(NB * 64 * 64 * 64 + 255) / 256, tb, 0, stream>>>(XB, d3w, d3b, XA, 128, 64, 64, 64);
  k_conv3<<<(NB * 64 * 64 * 64 + 255) / 256, tb, 0, stream>>>(XA, d4w, d4b, XB, 64, 64, 64, 64);
  k_up2<<<(NB * 64 * 128 * 128 + 255) / 256, tb, 0, stream>>>(XB, XA, 64, 64, 64);
  k_conv3<<<(NB * 3 * 128 * 128 + 255) / 256, tb, 0, stream>>>(XA, d5w, d5b, (float*)d_out, 64, 3, 128, 128);
}
